// Conductor_30966714204169
// MI455X (gfx1250) — matrix operand order repaired and checked by reading, not compiled, not run on hardware
//
#include <hip/hip_runtime.h>

#define T_STEPS 100
#define BATCH   32
#define NIN     512
#define NN      2048
#define AREAS   2
#define DTC     1e-3f
#define THR     1.0f
#define ALPHA   0.90483741803595957316f  // exp(-1e-3/0.01)

typedef __attribute__((ext_vector_type(16))) __bf16        v16bf;
typedef __attribute__((ext_vector_type(8)))  float         v8f;
typedef __attribute__((ext_vector_type(4)))  unsigned int  u32x4;

union Frag32B {
    u32x4 q[2];
    v16bf v;
};

__device__ __forceinline__ unsigned short f2bf(float f) {
    unsigned int u = __builtin_bit_cast(unsigned int, f);
    unsigned int r = 0x7FFFu + ((u >> 16) & 1u);   // round-to-nearest-even
    return (unsigned short)((u + r) >> 16);
}

// ---------------------------------------------------------------------------
// Kernel 1: Poisson spikes for all T at once; f32 to output, bf16 for WMMA.
__global__ void spike_kernel(const float* __restrict__ rates,
                             const float* __restrict__ noise,
                             float* __restrict__ xis_out,
                             unsigned short* __restrict__ xi16, int total) {
    int i = blockIdx.x * blockDim.x + threadIdx.x;
    if (i >= total) return;
    float s = (noise[i] < rates[i] * DTC) ? 1.0f : 0.0f;
    xis_out[i] = s;
    xi16[i] = (s != 0.0f) ? (unsigned short)0x3F80u : (unsigned short)0u;
}

// Kernel 2: zero membrane potential and Xd buffer 0.
__global__ void init_state(float* __restrict__ V,
                           unsigned short* __restrict__ xd0, int total) {
    int i = blockIdx.x * blockDim.x + threadIdx.x;
    if (i >= total) return;
    V[i] = 0.0f;
    xd0[i] = 0u;
}

// Kernel 3: Win [A][NIN][N] f32  ->  WinT [A][N][NIN] bf16  (transposed)
__global__ void conv_win(const float* __restrict__ Win,
                         unsigned short* __restrict__ winT) {
    int i = blockIdx.x * 256 + threadIdx.x;       // total = A*N*NIN
    int a = i / (NN * NIN);
    int r = i % (NN * NIN);
    int n = r / NIN;
    int k = r % NIN;
    winT[i] = f2bf(Win[(a * NIN + k) * NN + n]);
}

// Kernel 4: Wrec [s][a][k][n] f32  ->  WrecT [s][a][n][k] bf16  (transposed)
__global__ void conv_wrec(const float* __restrict__ Wrec,
                          unsigned short* __restrict__ wrecT) {
    int i  = blockIdx.x * 256 + threadIdx.x;      // total = A*A*N*N
    int sa = i / (NN * NN);
    int r  = i % (NN * NN);
    int n  = r / NN;
    int k  = r % NN;
    wrecT[i] = f2bf(Wrec[(long)sa * NN * NN + (long)k * NN + n]);
}

// ---------------------------------------------------------------------------
// Kernel 5: one LIF timestep. One wave per (area, 16-col n-tile); the wave
// owns the full M=32 batch: each B (weight) fragment is loaded ONCE and fed
// to TWO WMMAs (two 16x16 accumulators), halving L2 weight traffic.
// Grid = AREAS * (NN/16) = 256 blocks of 32 threads.
__global__ void __launch_bounds__(32)
step_kernel(const unsigned short* __restrict__ xi16_t,   // [B][NIN] bf16, this t
            const unsigned short* __restrict__ xd_read,  // [A][B][N] bf16
            unsigned short* __restrict__ xd_write,       // [A][B][N] bf16
            const unsigned short* __restrict__ winT,     // [A][N][NIN] bf16
            const unsigned short* __restrict__ wrecT,    // [A][A][N][N] bf16 (s,a,n,k)
            float* __restrict__ V,                       // [A][B][N] f32
            float* __restrict__ Sout,                    // spikes out (f32), area-major
            int t) {
    const int bid  = blockIdx.x;          // 0..255
    const int a    = bid >> 7;            // area
    const int nt   = bid & 127;           // neuron tile (0..127)
    const int lane = threadIdx.x;
    const int half = lane >> 4;
    const int l16  = lane & 15;

    const int bcol = nt * 16 + l16;       // output neuron column

    v8f c0 = {};   // batch rows 0..15
    v8f c1 = {};   // batch rows 16..31

    // A fragment: 16x32 bf16, row m = (lane&15) + rowBase;
    // per-lane K = half*8..+8 and 16+half*8..+8
    auto loadA = [&](const unsigned short* base, int rowBase, int ldk, int kc) -> v16bf {
        const unsigned short* p = base + (long)(rowBase + l16) * ldk + kc + half * 8;
        Frag32B f;
        f.q[0] = *reinterpret_cast<const u32x4*>(p);
        f.q[1] = *reinterpret_cast<const u32x4*>(p + 16);
        return f.v;
    };
    // B fragment: 32x16 bf16, col n = lane&15; K = half*8..+8 and 16+half*8..+8
    // (weights pre-transposed to [n][k])
    auto loadB = [&](const unsigned short* base, int ldk, int kc) -> v16bf {
        const unsigned short* p = base + (long)bcol * ldk + kc + half * 8;
        Frag32B f;
        f.q[0] = *reinterpret_cast<const u32x4*>(p);
        f.q[1] = *reinterpret_cast<const u32x4*>(p + 16);
        return f.v;
    };

    // 1) feedforward: Xi[B,NIN] x Win[a][NIN,N]
    {
        const unsigned short* Bs = winT + (long)a * NN * NIN;
        for (int kc = 0; kc < NIN; kc += 32) {
            v16bf bf  = loadB(Bs, NIN, kc);
            v16bf af0 = loadA(xi16_t, 0,  NIN, kc);
            v16bf af1 = loadA(xi16_t, 16, NIN, kc);
            c0 = __builtin_amdgcn_wmma_f32_16x16x32_bf16(false, af0, false, bf,
                                                         (short)0, c0, false, false);
            c1 = __builtin_amdgcn_wmma_f32_16x16x32_bf16(false, af1, false, bf,
                                                         (short)0, c1, false, false);
        }
    }
    // 2) recurrent: sum_s Xd[s][B,N] x Wrec[s][a][N,N]
    for (int s = 0; s < AREAS; ++s) {
        const unsigned short* As = xd_read + (long)s * BATCH * NN;
        const unsigned short* Bs = wrecT + (long)(s * AREAS + a) * NN * NN;
        for (int kc = 0; kc < NN; kc += 32) {
            v16bf bf  = loadB(Bs, NN, kc);
            v16bf af0 = loadA(As, 0,  NN, kc);
            v16bf af1 = loadA(As, 16, NN, kc);
            c0 = __builtin_amdgcn_wmma_f32_16x16x32_bf16(false, af0, false, bf,
                                                         (short)0, c0, false, false);
            c1 = __builtin_amdgcn_wmma_f32_16x16x32_bf16(false, af1, false, bf,
                                                         (short)0, c1, false, false);
        }
    }

    // LIF update. C/D layout: lane -> n = l16, VGPR r -> m = half*8 + r.
    #pragma unroll
    for (int mt = 0; mt < 2; ++mt) {
        #pragma unroll
        for (int r = 0; r < 8; ++r) {
            int   b   = mt * 16 + half * 8 + r;
            long  idx = ((long)a * BATCH + b) * NN + bcol;
            float I   = mt ? c1[r] : c0[r];
            float v   = V[idx];
            float xd  = xd_read[idx] ? 1.0f : 0.0f;
            float vn  = ALPHA * v * (1.0f - xd) + I;
            float sp  = (vn >= THR) ? 1.0f : 0.0f;
            V[idx] = vn;
            xd_write[idx] = (sp != 0.0f) ? (unsigned short)0x3F80u : (unsigned short)0u;
            Sout[(long)a * T_STEPS * BATCH * NN + ((long)t * BATCH + b) * NN + bcol] = sp;
        }
    }
}

// ---------------------------------------------------------------------------
extern "C" void kernel_launch(void* const* d_in, const int* in_sizes, int n_in,
                              void* d_out, int out_size, void* d_ws, size_t ws_size,
                              hipStream_t stream) {
    const float* rates = (const float*)d_in[0];   // [T,B,NIN]
    const float* noise = (const float*)d_in[1];   // [T,B,NIN]
    const float* Win   = (const float*)d_in[2];   // [A,NIN,N]
    const float* Wrec  = (const float*)d_in[3];   // [A,A,N,N]
    float* out = (float*)d_out;                   // Xis | S0 | S1

    // workspace carve-out (256B aligned)
    char*  ws  = (char*)d_ws;
    size_t off = 0;
    auto carve = [&](size_t bytes) -> char* {
        char* p = ws + off;
        off = (off + bytes + 255) & ~(size_t)255;
        return p;
    };
    unsigned short* xi16  = (unsigned short*)carve(sizeof(unsigned short) * (size_t)T_STEPS * BATCH * NIN);
    unsigned short* winT  = (unsigned short*)carve(sizeof(unsigned short) * (size_t)AREAS * NN * NIN);
    unsigned short* wrecT = (unsigned short*)carve(sizeof(unsigned short) * (size_t)AREAS * AREAS * NN * NN);
    float*          Vst   = (float*)carve(sizeof(float) * (size_t)AREAS * BATCH * NN);
    unsigned short* xd0   = (unsigned short*)carve(sizeof(unsigned short) * (size_t)AREAS * BATCH * NN);
    unsigned short* xd1   = (unsigned short*)carve(sizeof(unsigned short) * (size_t)AREAS * BATCH * NN);

    const int totSpk = T_STEPS * BATCH * NIN;
    spike_kernel<<<(totSpk + 255) / 256, 256, 0, stream>>>(rates, noise, out, xi16, totSpk);

    const int totState = AREAS * BATCH * NN;
    init_state<<<(totState + 255) / 256, 256, 0, stream>>>(Vst, xd0, totState);

    conv_win<<<(AREAS * NN * NIN) / 256, 256, 0, stream>>>(Win, winT);
    conv_wrec<<<(AREAS * AREAS * NN * NN) / 256, 256, 0, stream>>>(Wrec, wrecT);

    float* Sout = out + (size_t)T_STEPS * BATCH * NIN;
    for (int t = 0; t < T_STEPS; ++t) {
        const unsigned short* xr = (t & 1) ? xd1 : xd0;
        unsigned short*       xw = (t & 1) ? xd0 : xd1;
        step_kernel<<<AREAS * (NN / 16), 32, 0, stream>>>(
            xi16 + (size_t)t * BATCH * NIN, xr, xw, winT, wrecT, Vst, Sout, t);
    }
}
